// MusicVAE_38929583570981
// MI455X (gfx1250) — compile-verified, matrix operand order checked by reading, not run on hardware
//
#include <hip/hip_runtime.h>
#include <hip/hip_bf16.h>
#include <math.h>

// ---------------------------------------------------------------------------
// MusicVAE forward on gfx1250 (MI455X), compile-only tuned.
//  - All matmuls via v_wmma_f32_16x16x32_bf16 (bf16 in, f32 accum).
//  - Projection GEMMs: 32x32 macro-tiles (2x2 WMMA tiles), 2 loads/WMMA.
//  - Recurrent steps: fused Whh-GEMM + GRU gates; 32x16 macro-tile x 3 gates
//    (6 accumulators) -> 1.67 loads/WMMA and half the per-step L2 W-traffic.
//  - Encoder input projections hoisted out of the serial chain as one big
//    GEMM per (layer,dir) when workspace permits (deterministic ws_size test).
// ---------------------------------------------------------------------------

#define B_   256
#define T_   128
#define NC_  96
#define H_   512
#define LAT_ 512
#define G_   (3 * H_)
#define CPB_ 4
#define NB_  (T_ / CPB_)

typedef __attribute__((ext_vector_type(16))) __bf16 v16bf;
typedef __attribute__((ext_vector_type(8)))  __bf16 v8bf;
typedef __attribute__((ext_vector_type(8)))  float  v8f;

// One lane's 16 bf16 elements of a 16x32 operand tile (ISA 7.12.2 layout):
// K = [kb..kb+7] and [kb+16..kb+23], kb = 0 (lanes 0-15) / 8 (lanes 16-31).
__device__ __forceinline__ v16bf ldpair(const __bf16* __restrict__ row, int kb) {
  v8bf a0 = *(const v8bf*)(row + kb);
  v8bf a1 = *(const v8bf*)(row + kb + 16);
  v16bf r;
#pragma unroll
  for (int i = 0; i < 8; ++i) { r[i] = a0[i]; r[i + 8] = a1[i]; }
  return r;
}

__device__ __forceinline__ v8f wmma_bf(v16bf a, v16bf b, v8f c) {
  return __builtin_amdgcn_wmma_f32_16x16x32_bf16(false, a, false, b,
                                                 (short)0, c, false, false);
}

// C[m,n] = sum_k A[m,k]*W[n,k] + bias[n].  32x32 macro-tile per wave.
// M%32==0, N%32==0, K%32==0 guaranteed by caller.
__global__ void gemm_bf16_wmma(const __bf16* __restrict__ A, int sa,
                               const __bf16* __restrict__ W,
                               const float* __restrict__ bias,
                               float* __restrict__ C, int sc,
                               int M, int N, int K) {
  const int lane  = threadIdx.x & 31;
  const int wave  = blockIdx.x * (blockDim.x >> 5) + (threadIdx.x >> 5);
  const int nMac  = N >> 5;
  const int total = (M >> 5) * nMac;
  if (wave >= total) return;
  const int mT  = (wave / nMac) << 1;   // 16-row tile index
  const int nT  = (wave % nMac) << 1;   // 16-col tile index
  const int r16 = lane & 15;
  const int kb  = (lane >> 4) << 3;

  const __bf16* a0 = A + (size_t)(mT * 16 + r16) * sa;
  const __bf16* a1 = a0 + (size_t)16 * sa;
  const __bf16* w0 = W + (size_t)(nT * 16 + r16) * K;
  const __bf16* w1 = w0 + (size_t)16 * K;

  v8f c00 = {}, c01 = {}, c10 = {}, c11 = {};
  for (int k = 0; k < K; k += 32) {
    v16bf av0 = ldpair(a0 + k, kb);
    v16bf av1 = ldpair(a1 + k, kb);
    v16bf wv0 = ldpair(w0 + k, kb);
    v16bf wv1 = ldpair(w1 + k, kb);
    c00 = wmma_bf(av0, wv0, c00);
    c01 = wmma_bf(av0, wv1, c01);
    c10 = wmma_bf(av1, wv0, c10);
    c11 = wmma_bf(av1, wv1, c11);
  }
  const int rowOff = (lane >> 4) << 3;
  const float b0 = bias ? bias[nT * 16 + r16] : 0.0f;
  const float b1 = bias ? bias[(nT + 1) * 16 + r16] : 0.0f;
  float* cr0 = C + (size_t)(mT * 16 + rowOff) * sc + nT * 16 + r16;
  float* cr1 = cr0 + (size_t)16 * sc;
#pragma unroll
  for (int v = 0; v < 8; ++v) {
    cr0[(size_t)v * sc]      = c00[v] + b0;
    cr0[(size_t)v * sc + 16] = c01[v] + b1;
    cr1[(size_t)v * sc]      = c10[v] + b0;
    cr1[(size_t)v * sc + 16] = c11[v] + b1;
  }
}

// Fused GRU recurrent step.  Per wave: 32 batch rows x 16 hidden cols,
// 3 gate tiles (r/z/n) per row-half -> 6 accumulators sharing 2 A operands
// and 3 W operands per k-step.  Gates applied in the epilogue.
// h_prev buffers are ping-ponged by the caller (no read/write aliasing).
__global__ void gru_step_wmma(const __bf16* __restrict__ Hb,     // [B,H] bf16
                              const __bf16* __restrict__ Whh,    // [3H,H] bf16
                              const float* __restrict__ bhh,     // [3H]
                              const float* __restrict__ gx, long sgx,
                              const float* __restrict__ hprev,   // [B,H] f32
                              float* __restrict__ hout,          // [B,H] f32
                              __bf16* __restrict__ hbout, int shb,
                              __bf16* __restrict__ y2, long sy2) {
  const int lane = threadIdx.x & 31;
  const int wave = blockIdx.x * (blockDim.x >> 5) + (threadIdx.x >> 5);
  const int nTil = H_ >> 4;                       // 32
  if (wave >= (B_ >> 5) * nTil) return;
  const int mT  = (wave / nTil) << 1;             // 16-row tile index
  const int nT  = wave % nTil;
  const int r16 = lane & 15;
  const int kb  = (lane >> 4) << 3;

  const __bf16* aR0 = Hb + (size_t)(mT * 16 + r16) * H_;
  const __bf16* aR1 = aR0 + (size_t)16 * H_;
  const __bf16* wr  = Whh + (size_t)(nT * 16 + r16) * H_;
  const __bf16* wz  = wr + (size_t)H_ * H_;
  const __bf16* wn  = wz + (size_t)H_ * H_;

  v8f ar[2] = {}, az[2] = {}, an[2] = {};
  for (int k = 0; k < H_; k += 32) {
    v16bf av0 = ldpair(aR0 + k, kb);
    v16bf av1 = ldpair(aR1 + k, kb);
    v16bf wvr = ldpair(wr + k, kb);
    v16bf wvz = ldpair(wz + k, kb);
    v16bf wvn = ldpair(wn + k, kb);
    ar[0] = wmma_bf(av0, wvr, ar[0]);
    az[0] = wmma_bf(av0, wvz, az[0]);
    an[0] = wmma_bf(av0, wvn, an[0]);
    ar[1] = wmma_bf(av1, wvr, ar[1]);
    az[1] = wmma_bf(av1, wvz, az[1]);
    an[1] = wmma_bf(av1, wvn, an[1]);
  }
  const int j  = nT * 16 + r16;
  const float br = bhh[j], bz = bhh[H_ + j], bn = bhh[2 * H_ + j];
  const int rowOff = (lane >> 4) << 3;
#pragma unroll
  for (int mi = 0; mi < 2; ++mi) {
    const int b0 = (mT + mi) * 16 + rowOff;
#pragma unroll
    for (int v = 0; v < 8; ++v) {
      const int b = b0 + v;
      const float* gxr = gx + (size_t)b * sgx;
      float r  = 1.0f / (1.0f + __expf(-(gxr[j] + ar[mi][v] + br)));
      float z  = 1.0f / (1.0f + __expf(-(gxr[H_ + j] + az[mi][v] + bz)));
      float n  = tanhf(gxr[2 * H_ + j] + r * (an[mi][v] + bn));
      float hn = (1.0f - z) * n + z * hprev[(size_t)b * H_ + j];
      hout[(size_t)b * H_ + j] = hn;
      __bf16 hv = (__bf16)hn;
      hbout[(size_t)b * shb + j] = hv;
      if (y2) y2[(size_t)b * sy2 + j] = hv;
    }
  }
}

__global__ void f32_to_bf16_k(const float* __restrict__ s,
                              __bf16* __restrict__ d, int n) {
  int i = blockIdx.x * blockDim.x + threadIdx.x;
  if (i < n) d[i] = (__bf16)s[i];
}

__global__ void copy_bf_strided(const __bf16* __restrict__ s, int ss,
                                __bf16* __restrict__ d, int ds,
                                int rows, int cols) {
  int i = blockIdx.x * blockDim.x + threadIdx.x;
  if (i >= rows * cols) return;
  int r = i / cols, c = i - r * cols;
  d[(size_t)r * ds + c] = s[(size_t)r * ss + c];
}

__global__ void make_z(const float* __restrict__ mu, const float* __restrict__ lv,
                       const float* __restrict__ eps, __bf16* __restrict__ zb, int n) {
  int i = blockIdx.x * blockDim.x + threadIdx.x;
  if (i < n) zb[i] = (__bf16)(eps[i] * __expf(0.5f * lv[i]) + mu[i]);
}

__global__ void pack_decin(const __bf16* __restrict__ h1c,
                           const __bf16* __restrict__ xb,
                           int bar, __bf16* __restrict__ decin) {
  int i = blockIdx.x * blockDim.x + threadIdx.x;
  const int D = H_ + NC_;
  if (i >= B_ * CPB_ * D) return;
  int j  = i % D;
  int bt = i / D;
  int t  = bt & 3;
  int b  = bt >> 2;
  __bf16 v = (j < H_) ? h1c[(size_t)b * H_ + j]
                      : xb[((size_t)b * T_ + bar * CPB_ + t) * NC_ + (j - H_)];
  decin[i] = v;
}

__global__ void softmax96(const float* __restrict__ logits,
                          float* __restrict__ sm, float* __restrict__ lsm) {
  int row = blockIdx.x * (blockDim.x >> 5) + (threadIdx.x >> 5);
  if (row >= B_ * T_) return;
  int lane = threadIdx.x & 31;
  const float* x = logits + (size_t)row * NC_;
  float v0 = x[lane], v1 = x[lane + 32], v2 = x[lane + 64];
  float m = fmaxf(v0, fmaxf(v1, v2));
#pragma unroll
  for (int o = 16; o >= 1; o >>= 1) m = fmaxf(m, __shfl_xor(m, o, 32));
  float e0 = __expf(v0 - m), e1 = __expf(v1 - m), e2 = __expf(v2 - m);
  float s = e0 + e1 + e2;
#pragma unroll
  for (int o = 16; o >= 1; o >>= 1) s += __shfl_xor(s, o, 32);
  float inv = 1.0f / s, ls = logf(s);
  size_t base = (size_t)row * NC_;
  sm[base + lane]      = e0 * inv;
  sm[base + lane + 32] = e1 * inv;
  sm[base + lane + 64] = e2 * inv;
  lsm[base + lane]      = (v0 - m) - ls;
  lsm[base + lane + 32] = (v1 - m) - ls;
  lsm[base + lane + 64] = (v2 - m) - ls;
}

// ---------------------------------------------------------------------------

extern "C" void kernel_launch(void* const* d_in, const int* in_sizes, int n_in,
                              void* d_out, int out_size, void* d_ws, size_t ws_size,
                              hipStream_t stream) {
  (void)in_sizes; (void)n_in; (void)out_size;

  const float* x_f        = (const float*)d_in[0];
  const float* eps        = (const float*)d_in[2];
  const float* enc_bih[2] = { (const float*)d_in[5],  (const float*)d_in[9]  };
  const float* enc_bhh[2] = { (const float*)d_in[6],  (const float*)d_in[10] };
  const float* con0_bih = (const float*)d_in[13];
  const float* con0_bhh = (const float*)d_in[14];
  const float* con1_bih = (const float*)d_in[17];
  const float* con1_bhh = (const float*)d_in[18];
  const float* dec0_bih = (const float*)d_in[21];
  const float* dec0_bhh = (const float*)d_in[22];
  const float* dec1_bih = (const float*)d_in[25];
  const float* dec1_bhh = (const float*)d_in[26];
  const float* b_mu  = (const float*)d_in[28];
  const float* b_lv  = (const float*)d_in[30];
  const float* b_ci  = (const float*)d_in[32];
  const float* b_ch  = (const float*)d_in[34];
  const float* b_out = (const float*)d_in[36];

  float* out = (float*)d_out;
  const size_t OFF_SM  = 0;
  const size_t OFF_LSM = (size_t)B_ * T_ * NC_;
  const size_t OFF_MU  = 2 * OFF_LSM;
  const size_t OFF_LV  = OFF_MU + (size_t)B_ * LAT_;
  const size_t OFF_IN  = OFF_LV + (size_t)B_ * LAT_;

  size_t off = 0;
  auto alloc = [&](size_t bytes) -> char* {
    char* p = (char*)d_ws + off;
    off = (off + bytes + 255) & ~(size_t)255;
    return p;
  };
  auto allocF = [&](size_t n) { return (float*)alloc(n * sizeof(float)); };
  auto allocB = [&](size_t n) { return (__bf16*)alloc(n * sizeof(__bf16)); };

  auto cvt = [&](const void* src, size_t n) -> __bf16* {
    __bf16* d = allocB(n);
    f32_to_bf16_k<<<(int)((n + 255) / 256), 256, 0, stream>>>((const float*)src, d, (int)n);
    return d;
  };
  auto gemm = [&](const __bf16* A, int sa, const __bf16* W, const float* bias,
                  float* C, int sc, int M, int N, int K) {
    int tiles  = (M / 32) * (N / 32);
    int blocks = (tiles + 7) / 8;                 // 8 waves / 256-thread block
    gemm_bf16_wmma<<<blocks, 256, 0, stream>>>(A, sa, W, bias, C, sc, M, N, K);
  };
  auto step = [&](const __bf16* Hb, const __bf16* Whh, const float* bhh,
                  const float* gx, long sgx, const float* hprev,
                  float* hout, __bf16* hbout, __bf16* y2, long sy2) {
    // (B/32)*(H/16) = 256 waves -> 32 blocks of 8 waves
    gru_step_wmma<<<32, 256, 0, stream>>>(Hb, Whh, bhh, gx, sgx,
                                          hprev, hout, hbout, H_, y2, sy2);
  };

  // ---- bf16 weight mirrors (live in L2 afterwards) ----
  __bf16* enc0Wih_b = cvt(d_in[3],  (size_t)2 * G_ * NC_);
  __bf16* enc0Whh_b = cvt(d_in[4],  (size_t)2 * G_ * H_);
  __bf16* enc1Wih_b = cvt(d_in[7],  (size_t)2 * G_ * 2 * H_);
  __bf16* enc1Whh_b = cvt(d_in[8],  (size_t)2 * G_ * H_);
  __bf16* con0Wih_b = cvt(d_in[11], (size_t)G_ * NC_);
  __bf16* con0Whh_b = cvt(d_in[12], (size_t)G_ * H_);
  __bf16* con1Wih_b = cvt(d_in[15], (size_t)G_ * H_);
  __bf16* con1Whh_b = cvt(d_in[16], (size_t)G_ * H_);
  __bf16* dec0Wih_b = cvt(d_in[19], (size_t)G_ * (H_ + NC_));
  __bf16* dec0Whh_b = cvt(d_in[20], (size_t)G_ * H_);
  __bf16* dec1Wih_b = cvt(d_in[23], (size_t)G_ * H_);
  __bf16* dec1Whh_b = cvt(d_in[24], (size_t)G_ * H_);
  __bf16* Wmu_b     = cvt(d_in[27], (size_t)LAT_ * 4 * H_);
  __bf16* Wlv_b     = cvt(d_in[29], (size_t)LAT_ * 4 * H_);
  __bf16* Wci_b     = cvt(d_in[31], (size_t)NC_ * LAT_);
  __bf16* Wch_b     = cvt(d_in[33], (size_t)H_ * LAT_);
  __bf16* Wout_b    = cvt(d_in[35], (size_t)NC_ * H_);

  // ---- activations ----
  __bf16* xb   = cvt(x_f, (size_t)B_ * T_ * NC_);           // [B,T,NC]
  __bf16* y0   = allocB((size_t)B_ * T_ * 2 * H_);          // [B,T,2H]
  float*  gx_s = allocF((size_t)B_ * G_);                   // small gx scratch
  float*  hEf[4][2]; __bf16* hEb[4][2];                     // ping-pong states
  for (int c = 0; c < 4; ++c)
    for (int p = 0; p < 2; ++p) {
      hEf[c][p] = allocF((size_t)B_ * H_);
      hEb[c][p] = allocB((size_t)B_ * H_);
    }

  // decoder buffers
  __bf16* hidden = allocB((size_t)B_ * 4 * H_);
  __bf16* zb     = allocB((size_t)B_ * LAT_);
  float*  ci_f   = allocF((size_t)B_ * NC_);
  float*  gxc0   = allocF((size_t)B_ * G_);
  float*  h0cF[2]; __bf16* h0cB[2]; float* h1cF[2]; __bf16* h1cB[2];
  for (int p = 0; p < 2; ++p) {
    h0cF[p] = allocF((size_t)B_ * H_); h0cB[p] = allocB((size_t)B_ * H_);
    h1cF[p] = allocF((size_t)B_ * H_); h1cB[p] = allocB((size_t)B_ * H_);
  }
  __bf16* decin = allocB((size_t)B_ * CPB_ * (H_ + NC_));
  float*  gx0   = allocF((size_t)B_ * CPB_ * G_);
  float*  gx1   = allocF((size_t)B_ * CPB_ * G_);
  float*  hdF[2][2]; __bf16* hdB[2][2];
  for (int l = 0; l < 2; ++l)
    for (int p = 0; p < 2; ++p) {
      hdF[l][p] = allocF((size_t)B_ * H_); hdB[l][p] = allocB((size_t)B_ * H_);
    }
  __bf16* d0y    = allocB((size_t)B_ * CPB_ * H_);
  __bf16* d1y    = allocB((size_t)B_ * CPB_ * H_);
  float*  logits = allocF((size_t)B_ * T_ * NC_);

  // Big path: encoder gx for all timesteps as one GEMM per (layer,dir).
  // Deterministic branch on ws_size (same every call).
  const size_t gxAllBytes = (size_t)B_ * T_ * G_ * sizeof(float);
  float* gx_all = nullptr;
  if ((long long)ws_size - (long long)off >=
      (long long)(gxAllBytes + (8 << 20)))
    gx_all = allocF((size_t)B_ * T_ * G_);
  const bool big = (gx_all != nullptr);

  // ---- encoder: 2 layers x 2 directions ----
  const __bf16* Wih_l[2] = { enc0Wih_b, enc1Wih_b };
  const __bf16* Whh_l[2] = { enc0Whh_b, enc1Whh_b };
  const int     Din_l[2] = { NC_, 2 * H_ };
  for (int layer = 0; layer < 2; ++layer) {
    for (int dir = 0; dir < 2; ++dir) {
      const int c = layer * 2 + dir;
      hipMemsetAsync(hEf[c][0], 0, (size_t)B_ * H_ * sizeof(float), stream);
      hipMemsetAsync(hEb[c][0], 0, (size_t)B_ * H_ * sizeof(__bf16), stream);
      const int Din = Din_l[layer];
      const __bf16* Wih = Wih_l[layer] + (size_t)dir * G_ * Din;
      const __bf16* Whh = Whh_l[layer] + (size_t)dir * G_ * H_;
      const float*  bih = enc_bih[layer] + (size_t)dir * G_;
      const float*  bhh = enc_bhh[layer] + (size_t)dir * G_;
      const __bf16* Xin = layer ? y0 : xb;
      const int     sxB = layer ? T_ * 2 * H_ : T_ * NC_;   // per-batch stride
      if (big)  // rows m = b*T+t are contiguous with stride Din
        gemm(Xin, Din, Wih, bih, gx_all, G_, B_ * T_, G_, Din);
      int cur = 0;
      for (int s = 0; s < T_; ++s) {
        const int t = dir ? (T_ - 1 - s) : s;
        const float* gxp; long sgx;
        if (big) { gxp = gx_all + (size_t)t * G_; sgx = (long)T_ * G_; }
        else {
          gemm(Xin + (size_t)t * Din, sxB, Wih, bih, gx_s, G_, B_, G_, Din);
          gxp = gx_s; sgx = G_;
        }
        __bf16* y2 = (layer == 0)
                       ? y0 + (size_t)t * 2 * H_ + (size_t)dir * H_ : nullptr;
        step(hEb[c][cur], Whh, bhh, gxp, sgx, hEf[c][cur],
             hEf[c][cur ^ 1], hEb[c][cur ^ 1], y2, (long)T_ * 2 * H_);
        cur ^= 1;
      }
      // T_=128 even => after the loop the final state sits in slot 0
    }
  }

  // ---- latent ----
  for (int c = 0; c < 4; ++c)
    copy_bf_strided<<<(B_ * H_ + 255) / 256, 256, 0, stream>>>(
        hEb[c][0], H_, hidden + (size_t)c * H_, 4 * H_, B_, H_);
  gemm(hidden, 4 * H_, Wmu_b, b_mu, out + OFF_MU, LAT_, B_, LAT_, 4 * H_);
  gemm(hidden, 4 * H_, Wlv_b, b_lv, out + OFF_LV, LAT_, B_, LAT_, 4 * H_);
  make_z<<<(B_ * LAT_ + 255) / 256, 256, 0, stream>>>(out + OFF_MU, out + OFF_LV,
                                                      eps, zb, B_ * LAT_);

  // ---- decoder setup ----
  gemm(zb, LAT_, Wci_b, b_ci, ci_f, NC_, B_, NC_, LAT_);
  __bf16* ci_b = cvt(ci_f, (size_t)B_ * NC_);
  gemm(ci_b, NC_, con0Wih_b, con0_bih, gxc0, G_, B_, G_, NC_);
  gemm(zb, LAT_, Wch_b, b_ch, h0cF[0], H_, B_, H_, LAT_);
  f32_to_bf16_k<<<(B_ * H_ + 255) / 256, 256, 0, stream>>>(h0cF[0], h0cB[0], B_ * H_);
  hipMemcpyAsync(h1cF[0], h0cF[0], (size_t)B_ * H_ * sizeof(float),
                 hipMemcpyDeviceToDevice, stream);
  hipMemcpyAsync(h1cB[0], h0cB[0], (size_t)B_ * H_ * sizeof(__bf16),
                 hipMemcpyDeviceToDevice, stream);

  // ---- decoder: 32 bars ----
  int c0 = 0, c1 = 0;
  for (int bar = 0; bar < NB_; ++bar) {
    // conductor layer 0: h0c = GRUcell(h0c, gx_c0)
    step(h0cB[c0], con0Whh_b, con0_bhh, gxc0, G_, h0cF[c0],
         h0cF[c0 ^ 1], h0cB[c0 ^ 1], nullptr, 0);
    c0 ^= 1;
    // conductor layer 1: h1c = GRUcell(h1c, h0c@Wih + bih)
    gemm(h0cB[c0], H_, con1Wih_b, con1_bih, gx_s, G_, B_, G_, H_);
    step(h1cB[c1], con1Whh_b, con1_bhh, gx_s, G_, h1cF[c1],
         h1cF[c1 ^ 1], h1cB[c1 ^ 1], nullptr, 0);
    c1 ^= 1;
    // bar inputs + layer-0 input projection for all 4 chords
    pack_decin<<<(B_ * CPB_ * (H_ + NC_) + 255) / 256, 256, 0, stream>>>(
        h1cB[c1], xb, bar, decin);
    gemm(decin, H_ + NC_, dec0Wih_b, dec0_bih, gx0, G_, B_ * CPB_, G_, H_ + NC_);
    int p = 0;
    for (int t = 0; t < CPB_; ++t) {
      const __bf16* hpB = t ? hdB[0][p] : h0cB[c0];
      const float*  hpF = t ? hdF[0][p] : h0cF[c0];
      step(hpB, dec0Whh_b, dec0_bhh, gx0 + (size_t)t * G_, (long)CPB_ * G_,
           hpF, hdF[0][p ^ 1], hdB[0][p ^ 1], d0y + (size_t)t * H_, (long)CPB_ * H_);
      p ^= 1;
    }
    gemm(d0y, H_, dec1Wih_b, dec1_bih, gx1, G_, B_ * CPB_, G_, H_);
    p = 0;
    for (int t = 0; t < CPB_; ++t) {
      const __bf16* hpB = t ? hdB[1][p] : h1cB[c1];
      const float*  hpF = t ? hdF[1][p] : h1cF[c1];
      step(hpB, dec1Whh_b, dec1_bhh, gx1 + (size_t)t * G_, (long)CPB_ * G_,
           hpF, hdF[1][p ^ 1], hdB[1][p ^ 1], d1y + (size_t)t * H_, (long)CPB_ * H_);
      p ^= 1;
    }
    for (int t = 0; t < CPB_; ++t)
      gemm(d1y + (size_t)t * H_, CPB_ * H_, Wout_b, b_out,
           logits + (size_t)(bar * CPB_ + t) * NC_, T_ * NC_, B_, NC_, H_);
  }

  // ---- outputs ----
  softmax96<<<(B_ * T_ + 3) / 4, 128, 0, stream>>>(logits, out + OFF_SM, out + OFF_LSM);
  hipMemcpyAsync(out + OFF_IN, x_f, (size_t)B_ * T_ * NC_ * sizeof(float),
                 hipMemcpyDeviceToDevice, stream);
}
